// SpatialCrossAttentionBlock_30958124269669
// MI455X (gfx1250) — matrix operand order repaired and checked by reading, not compiled, not run on hardware
//
#include <hip/hip_runtime.h>
#include <hip/hip_bf16.h>
#include <cstdint>

// ---------------------------------------------------------------------------
// SpatialCrossAttentionBlock for MI455X (gfx1250, wave32, WMMA f16 path)
// B=8, C=512, H=W=32 (N=1024), C_CTX=256, HEADS=8, HEAD_DIM=64, GROUPS=32
// All GEMM B-operands staged via LDS in WMMA fragment order (2x ds_load_b128
// per lane); attention operands stored so every fragment has unit k-stride.
// ---------------------------------------------------------------------------

#define BB 8
#define CC 512
#define NN 1024            // H*W
#define CCTX 256
#define HEADS 8
#define HDIM 64
#define C2 1024            // 2*C
#define C3 1536            // 3*C

typedef __attribute__((ext_vector_type(16))) _Float16 v16h;
typedef __attribute__((ext_vector_type(8)))  float    v8f;

// ------------------------- WMMA fragment helpers ---------------------------
// A-matrix 16x32 f16 (MxK): lane l -> row m=l&15, half h=l>>4,
// element e -> K = e + 16*h
template <typename T>
__device__ __forceinline__ v16h load_frag_a(const T* p, int sm, int sk) {
  const int l = threadIdx.x & 31;
  const int m = l & 15, hh = l >> 4;
  v16h a;
#pragma unroll
  for (int e = 0; e < 16; ++e) {
    const int k = e + (hh << 4);
    a[e] = (_Float16)p[m * sm + k * sk];
  }
  return a;
}

// B-matrix 32x16 f16 (KxN): lane l -> col n=l&15, half h=l>>4,
// element e -> K = e + 16*h
template <typename T>
__device__ __forceinline__ v16h load_frag_b(const T* p, int sk, int sn) {
  const int l = threadIdx.x & 31;
  const int n = l & 15, hh = l >> 4;
  v16h b;
#pragma unroll
  for (int e = 0; e < 16; ++e) {
    const int k = e + (hh << 4);
    b[e] = (_Float16)p[k * sk + n * sn];
  }
  return b;
}

__device__ __forceinline__ v8f wmma_f16(v16h a, v16h b, v8f c) {
  return __builtin_amdgcn_wmma_f32_16x16x32_f16(
      false, a, false, b, (short)0, c, false, false);
}

__device__ __forceinline__ v8f vzero8() {
  v8f z = {0.f, 0.f, 0.f, 0.f, 0.f, 0.f, 0.f, 0.f};
  return z;
}

// Stage a 32k x 32n B tile (row-major, leading dim ldn) into LDS arranged in
// exact WMMA-B fragment order: ldsB[(strip*32 + lane)*16 + e].
// Global side: each of 256 threads loads 4 contiguous halfs (coalesced b64).
__device__ __forceinline__ void stage_b_32x32(const _Float16* __restrict__ gB,
                                              int ldn, _Float16* ldsB) {
  const int t = threadIdx.x;
  const int k = t >> 3, nq = (t & 7) << 2;
  const _Float16* gp = gB + (size_t)k * ldn + nq;
  _Float16 v0 = gp[0], v1 = gp[1], v2 = gp[2], v3 = gp[3];
#pragma unroll
  for (int j = 0; j < 4; ++j) {
    _Float16 v = (j == 0) ? v0 : (j == 1) ? v1 : (j == 2) ? v2 : v3;
    const int n = nq + j;
    const int s = n >> 4;
    const int ll = (n & 15) + ((k >> 4) << 4);
    const int e = k & 15;
    ldsB[((s << 5) + ll) * 16 + e] = v;
  }
}

// --------------------------- small converters ------------------------------
__global__ void k_f32_to_f16(const float* __restrict__ s,
                             _Float16* __restrict__ d, int n) {
  int i = blockIdx.x * 256 + threadIdx.x;
  if (i < n) d[i] = (_Float16)s[i];
}

// ------------------- pool 64x64 -> 32x32 + GroupNorm(ctx) ------------------
__global__ void k_pool_gn_ctx(const float* __restrict__ ctx,
                              const float* __restrict__ gw,
                              const float* __restrict__ gb,
                              _Float16* __restrict__ ctxn) {
  __shared__ float pool[8192];
  __shared__ float red[256], red2[256];
  __shared__ float s_mean, s_rstd;
  const int b = blockIdx.x >> 5, g = blockIdx.x & 31;
  const int t = threadIdx.x;
  float s = 0.f, s2 = 0.f;
#pragma unroll 4
  for (int i = 0; i < 32; ++i) {
    int idx = t + (i << 8);                 // 0..8191 over (8ch x 1024)
    int cl = idx >> 10, n = idx & 1023;
    int pi = n >> 5, pj = n & 31;
    const float* p =
        ctx + ((((size_t)b * CCTX + g * 8 + cl) * 64 + 2 * pi) * 64 + 2 * pj);
    float v = 0.25f * (p[0] + p[1] + p[64] + p[65]);
    pool[idx] = v;
    s += v;
    s2 += v * v;
  }
  red[t] = s; red2[t] = s2;
  __syncthreads();
  for (int o = 128; o > 0; o >>= 1) {
    if (t < o) { red[t] += red[t + o]; red2[t] += red2[t + o]; }
    __syncthreads();
  }
  if (t == 0) {
    float m = red[0] * (1.f / 8192.f);
    float var = red2[0] * (1.f / 8192.f) - m * m;
    s_mean = m;
    s_rstd = rsqrtf(var + 1e-5f);
  }
  __syncthreads();
  const float mean = s_mean, rstd = s_rstd;
#pragma unroll 4
  for (int i = 0; i < 32; ++i) {
    int idx = t + (i << 8);
    int cl = idx >> 10, n = idx & 1023;
    int c = g * 8 + cl;
    float v = (pool[idx] - mean) * rstd * gw[c] + gb[c];
    ctxn[((size_t)b * CCTX + c) * NN + n] = (_Float16)v;
  }
}

// ---------- ctx 1x1 conv: [512x256]@[256x1024]; block = 128m x 32n ---------
__global__ __launch_bounds__(256) void k_gemm_ctx(
    const _Float16* __restrict__ w, const _Float16* __restrict__ ctxn,
    const float* __restrict__ bias, float* __restrict__ ctxp) {
  __shared__ v16h ldsBv[64];                // 2 strips x 32 lanes, 2 KB
  _Float16* ldsB = reinterpret_cast<_Float16*>(ldsBv);
  const int b = blockIdx.x >> 7;            // 4*32 blocks per batch
  const int r = blockIdx.x & 127;
  const int m0 = ((r >> 5) << 7) + ((threadIdx.x >> 5) << 4);
  const int n0 = (r & 31) << 5;
  const _Float16* bb = ctxn + ((size_t)b * CCTX) * NN + n0;
  const int l = threadIdx.x & 31;
  v8f acc0 = vzero8(), acc1 = vzero8();
  for (int k0 = 0; k0 < CCTX; k0 += 32) {
    if (k0 + 32 < CCTX)
      __builtin_prefetch(bb + (size_t)(k0 + 32) * NN + (threadIdx.x >> 3), 0, 0);
    stage_b_32x32(bb + (size_t)k0 * NN, NN, ldsB);
    __syncthreads();
    v16h a = load_frag_a(w + (size_t)m0 * CCTX + k0, CCTX, 1);
    v16h b0 = ldsBv[l];
    v16h b1 = ldsBv[32 + l];
    acc0 = wmma_f16(a, b0, acc0);
    acc1 = wmma_f16(a, b1, acc1);
    __syncthreads();
  }
  const int nl = l & 15, hh = l >> 4;
#pragma unroll
  for (int s = 0; s < 2; ++s) {
    v8f acc = s ? acc1 : acc0;
    float* dst = ctxp + ((size_t)b * CC + m0 + (hh << 3)) * NN + n0 + (s << 4) + nl;
#pragma unroll
    for (int e = 0; e < 8; ++e)
      dst[(size_t)e * NN] = acc[e] + bias[m0 + (hh << 3) + e];
  }
}

// ----------- GroupNorm over concat [x | ctx_proj] (1024 channels) ----------
__global__ void k_gn_concat(const float* __restrict__ x,
                            const float* __restrict__ ctxp,
                            const float* __restrict__ gw,
                            const float* __restrict__ gb,
                            _Float16* __restrict__ xn) {
  __shared__ float red[256], red2[256];
  __shared__ float s_mean, s_rstd;
  const int b = blockIdx.x >> 5, g = blockIdx.x & 31;
  const int t = threadIdx.x;
  float s = 0.f, s2 = 0.f;
  for (int i = 0; i < 128; ++i) {
    int idx = t + (i << 8);             // 0..32767 over (32ch x 1024)
    int cl = idx >> 10, n = idx & 1023;
    int gc = g * 32 + cl;
    float v = (gc < CC) ? x[((size_t)b * CC + gc) * NN + n]
                        : ctxp[((size_t)b * CC + (gc - CC)) * NN + n];
    s += v;
    s2 += v * v;
  }
  red[t] = s; red2[t] = s2;
  __syncthreads();
  for (int o = 128; o > 0; o >>= 1) {
    if (t < o) { red[t] += red[t + o]; red2[t] += red2[t + o]; }
    __syncthreads();
  }
  if (t == 0) {
    float m = red[0] * (1.f / 32768.f);
    float var = red2[0] * (1.f / 32768.f) - m * m;
    s_mean = m;
    s_rstd = rsqrtf(var + 1e-5f);
  }
  __syncthreads();
  const float mean = s_mean, rstd = s_rstd;
  for (int i = 0; i < 128; ++i) {
    int idx = t + (i << 8);
    int cl = idx >> 10, n = idx & 1023;
    int gc = g * 32 + cl;
    float v = (gc < CC) ? x[((size_t)b * CC + gc) * NN + n]
                        : ctxp[((size_t)b * CC + (gc - CC)) * NN + n];
    float vn = (v - mean) * rstd * gw[gc] + gb[gc];
    xn[((size_t)b * C2 + gc) * NN + n] = (_Float16)vn;
  }
}

// -------- qkv: [1536x1024]@[1024x1024]; writes Q,K as [b,h,n,d] ------------
// (transposed store is per-lane contiguous -> b128), V stays channel-major.
__global__ __launch_bounds__(256) void k_gemm_qkv(
    const _Float16* __restrict__ w, const _Float16* __restrict__ xn,
    const float* __restrict__ bias, _Float16* __restrict__ qt,
    _Float16* __restrict__ kt, _Float16* __restrict__ qv) {
  __shared__ v16h ldsBv[64];
  _Float16* ldsB = reinterpret_cast<_Float16*>(ldsBv);
  const int b = blockIdx.x / 384;           // 12*32 blocks per batch
  const int r = blockIdx.x % 384;
  const int m0 = ((r >> 5) << 7) + ((threadIdx.x >> 5) << 4);
  const int n0 = (r & 31) << 5;
  const _Float16* bb = xn + ((size_t)b * C2) * NN + n0;
  const int l = threadIdx.x & 31;
  v8f acc0 = vzero8(), acc1 = vzero8();
  for (int k0 = 0; k0 < C2; k0 += 32) {
    if (k0 + 32 < C2)
      __builtin_prefetch(bb + (size_t)(k0 + 32) * NN + (threadIdx.x >> 3), 0, 0);
    stage_b_32x32(bb + (size_t)k0 * NN, NN, ldsB);
    __syncthreads();
    v16h a = load_frag_a(w + (size_t)m0 * C2 + k0, C2, 1);
    v16h b0 = ldsBv[l];
    v16h b1 = ldsBv[32 + l];
    acc0 = wmma_f16(a, b0, acc0);
    acc1 = wmma_f16(a, b1, acc1);
    __syncthreads();
  }
  const int nl = l & 15, hh = l >> 4;
#pragma unroll
  for (int s = 0; s < 2; ++s) {
    v8f acc = s ? acc1 : acc0;
    const int nn = n0 + (s << 4) + nl;
    if (m0 < CC) {                          // Q -> [b,h,n,d]
      const int hd = m0 >> 6, d0 = (m0 & 63) + (hh << 3);
      _Float16* dst = qt + (((size_t)b * HEADS + hd) * NN + nn) * HDIM + d0;
#pragma unroll
      for (int e = 0; e < 8; ++e)
        dst[e] = (_Float16)(acc[e] + bias[m0 + (hh << 3) + e]);
    } else if (m0 < C2) {                   // K -> [b,h,n,d]
      const int mm = m0 - CC;
      const int hd = mm >> 6, d0 = (mm & 63) + (hh << 3);
      _Float16* dst = kt + (((size_t)b * HEADS + hd) * NN + nn) * HDIM + d0;
#pragma unroll
      for (int e = 0; e < 8; ++e)
        dst[e] = (_Float16)(acc[e] + bias[m0 + (hh << 3) + e]);
    } else {                                // V -> [c,n] channel-major
      const int mm = m0 - C2;
      _Float16* dst = qv + ((size_t)b * CC + mm + (hh << 3)) * NN + nn;
#pragma unroll
      for (int e = 0; e < 8; ++e)
        dst[(size_t)e * NN] = (_Float16)(acc[e] + bias[m0 + (hh << 3) + e]);
    }
  }
}

// -------------------------------- attention --------------------------------
// 2 waves per block share 16 query rows and a 64 KB LDS score buffer.
// All fragments have unit k-stride per lane -> vectorized b128 loads.
__global__ __launch_bounds__(64) void k_attn(const _Float16* __restrict__ qt,
                                             const _Float16* __restrict__ kt,
                                             const _Float16* __restrict__ qv,
                                             _Float16* __restrict__ ao) {
  __shared__ float sc[16 * NN];             // 64 KB
  const int bh = blockIdx.x >> 6, rb = blockIdx.x & 63;
  const int b = bh >> 3, hd = bh & 7;
  const int n0 = rb << 4;
  const int wv = threadIdx.x >> 5;
  const int l = threadIdx.x & 31, nl = l & 15, hh = l >> 4;
  const _Float16* qb = qt + ((size_t)b * HEADS + hd) * NN * HDIM;
  const _Float16* kb = kt + ((size_t)b * HEADS + hd) * NN * HDIM;
  const _Float16* vb = qv + ((size_t)b * CC + hd * HDIM) * NN;

  // phase 1: S = Q[n,d] @ K[m,d]^T * scale  (wave wv: key tiles [32wv,32wv+32))
  v16h aq0 = load_frag_a(qb + (size_t)n0 * HDIM, HDIM, 1);
  v16h aq1 = load_frag_a(qb + (size_t)n0 * HDIM + 32, HDIM, 1);
  for (int jt = wv * 32; jt < wv * 32 + 32; ++jt) {
    const int n1 = jt << 4;
    v16h b0 = load_frag_b(kb + (size_t)n1 * HDIM, 1, HDIM);
    v16h b1 = load_frag_b(kb + (size_t)n1 * HDIM + 32, 1, HDIM);
    v8f acc = vzero8();
    acc = wmma_f16(aq0, b0, acc);
    acc = wmma_f16(aq1, b1, acc);
#pragma unroll
    for (int e = 0; e < 8; ++e)
      sc[(e + (hh << 3)) * NN + n1 + nl] = acc[e] * 0.125f;   // 1/sqrt(64)
  }
  __syncthreads();

  // phase 2: exact softmax; wave wv owns rows [8wv,8wv+8), 4 lanes per row
  {
    const int row = (wv << 3) + (l & 7);
    const int seg = l >> 3;                 // 0..3 -> 256 columns each
    float* rp = sc + row * NN + (seg << 8);
    float mx = -3.0e38f;
    for (int j = 0; j < 256; ++j) mx = fmaxf(mx, rp[j]);
    mx = fmaxf(mx, __shfl_xor(mx, 8, 32));
    mx = fmaxf(mx, __shfl_xor(mx, 16, 32));
    float sum = 0.f;
    for (int j = 0; j < 256; ++j) {
      float e = __expf(rp[j] - mx);
      rp[j] = e;
      sum += e;
    }
    sum += __shfl_xor(sum, 8, 32);
    sum += __shfl_xor(sum, 16, 32);
    const float inv = 1.0f / sum;
    for (int j = 0; j < 256; ++j) rp[j] *= inv;
  }
  __syncthreads();

  // phase 3: O[n,d] = P[n,m] @ V^T[m,d]; wave wv owns d-tiles {2wv, 2wv+1}
  v8f od0 = vzero8(), od1 = vzero8();
  const int d0a = wv << 5;                  // 0 or 32
  for (int jt = 0; jt < 32; ++jt) {
    const int k0 = jt << 5;
    v16h a = load_frag_a(sc + k0, NN, 1);   // f32 LDS -> f16 fragment
    v16h b0 = load_frag_b(vb + (size_t)d0a * NN + k0, 1, NN);
    v16h b1 = load_frag_b(vb + (size_t)(d0a + 16) * NN + k0, 1, NN);
    od0 = wmma_f16(a, b0, od0);
    od1 = wmma_f16(a, b1, od1);
  }
  _Float16* dst = ao + (((size_t)b * HEADS + hd) * NN + n0) * HDIM;
#pragma unroll
  for (int s = 0; s < 2; ++s) {
    v8f od = s ? od1 : od0;
#pragma unroll
    for (int e = 0; e < 8; ++e)
      dst[(e + (hh << 3)) * HDIM + d0a + (s << 4) + nl] = (_Float16)od[e];
  }
}

// ---------------- out proj [512x512] + bias + gate + residual --------------
// B (= attn out [b,h,n,d]) already has unit k-stride per lane.
__global__ __launch_bounds__(256) void k_out_proj(
    const _Float16* __restrict__ w, const _Float16* __restrict__ ao,
    const float* __restrict__ bias, const float* __restrict__ gate,
    const float* __restrict__ x, float* __restrict__ out) {
  const int wid = blockIdx.x * 8 + (threadIdx.x >> 5);
  const int b = wid >> 11;
  const int r = wid & 2047;
  const int m0 = (r >> 6) << 4;
  const int n0 = (r & 63) << 4;
  v8f acc = vzero8();
#pragma unroll
  for (int k0 = 0; k0 < CC; k0 += 32) {
    v16h a = load_frag_a(w + (size_t)m0 * CC + k0, CC, 1);
    // channel k0+k lives at ao[b][k0>>6][n][k0&63 + k]  (k<32 stays in-head)
    const _Float16* bb =
        ao + (((size_t)b * HEADS + (k0 >> 6)) * NN + n0) * HDIM + (k0 & 63);
    v16h bf = load_frag_b(bb, 1, HDIM);
    acc = wmma_f16(a, bf, acc);
  }
  const int l = threadIdx.x & 31, nl = l & 15, hh = l >> 4;
  const float gt = gate[b];
#pragma unroll
  for (int e = 0; e < 8; ++e) {
    int o = m0 + e + (hh << 3);
    size_t idx = ((size_t)b * CC + o) * NN + n0 + nl;
    out[idx] = x[idx] + gt * (acc[e] + bias[o]);
  }
}

// ------------------------------- launcher ----------------------------------
extern "C" void kernel_launch(void* const* d_in, const int* in_sizes, int n_in,
                              void* d_out, int out_size, void* d_ws,
                              size_t ws_size, hipStream_t stream) {
  (void)in_sizes; (void)n_in; (void)out_size; (void)ws_size;
  const float* x       = (const float*)d_in[0];
  const float* context = (const float*)d_in[1];
  const float* gate    = (const float*)d_in[2];
  const float* gn_ctx_w= (const float*)d_in[3];
  const float* gn_ctx_b= (const float*)d_in[4];
  const float* ctx_w   = (const float*)d_in[5];
  const float* ctx_b   = (const float*)d_in[6];
  const float* gn_w    = (const float*)d_in[7];
  const float* gn_b    = (const float*)d_in[8];
  const float* qkv_w   = (const float*)d_in[9];
  const float* qkv_b   = (const float*)d_in[10];
  const float* out_w   = (const float*)d_in[11];
  const float* out_b   = (const float*)d_in[12];
  float* outp = (float*)d_out;

  char* ws = (char*)d_ws;
  size_t off = 0;
  auto alloc = [&](size_t bytes) {
    char* p = ws + off;
    off = (off + bytes + 255) & ~size_t(255);
    return p;
  };
  _Float16* ctxn   = (_Float16*)alloc((size_t)BB * CCTX * NN * 2);         //  4 MB
  float*    ctxp   = (float*)   alloc((size_t)BB * CC * NN * 4);           // 16 MB
  _Float16* xn     = (_Float16*)alloc((size_t)BB * C2 * NN * 2);           // 16 MB
  _Float16* qt     = (_Float16*)alloc((size_t)BB * HEADS * NN * HDIM * 2); //  8 MB
  _Float16* kt     = (_Float16*)alloc((size_t)BB * HEADS * NN * HDIM * 2); //  8 MB
  _Float16* qv     = (_Float16*)alloc((size_t)BB * CC * NN * 2);           //  8 MB
  _Float16* ao     = (_Float16*)alloc((size_t)BB * HEADS * NN * HDIM * 2); //  8 MB
  _Float16* w_ctx16= (_Float16*)alloc((size_t)CC * CCTX * 2);
  _Float16* w_qkv16= (_Float16*)alloc((size_t)C3 * C2 * 2);
  _Float16* w_out16= (_Float16*)alloc((size_t)CC * CC * 2);

  // weight conversions (f32 -> f16)
  k_f32_to_f16<<<(CC * CCTX + 255) / 256, 256, 0, stream>>>(ctx_w, w_ctx16, CC * CCTX);
  k_f32_to_f16<<<(C3 * C2 + 255) / 256, 256, 0, stream>>>(qkv_w, w_qkv16, C3 * C2);
  k_f32_to_f16<<<(CC * CC + 255) / 256, 256, 0, stream>>>(out_w, w_out16, CC * CC);

  // pool + GroupNorm(ctx)
  k_pool_gn_ctx<<<BB * 32, 256, 0, stream>>>(context, gn_ctx_w, gn_ctx_b, ctxn);
  // ctx 1x1 conv (WMMA, LDS-staged B)
  k_gemm_ctx<<<BB * 4 * 32, 256, 0, stream>>>(w_ctx16, ctxn, ctx_b, ctxp);
  // GroupNorm over concat channels
  k_gn_concat<<<BB * 32, 256, 0, stream>>>(x, ctxp, gn_w, gn_b, xn);
  // qkv projection (WMMA, LDS-staged B, attention-friendly output layouts)
  k_gemm_qkv<<<BB * 12 * 32, 256, 0, stream>>>(w_qkv16, xn, qkv_b, qt, kt, qv);
  // attention (WMMA scores + WMMA PV, LDS-resident softmax, 2 waves/block)
  k_attn<<<BB * HEADS * (NN / 16), 64, 0, stream>>>(qt, kt, qv, ao);
  // out projection + gate + residual (WMMA)
  k_out_proj<<<BB * 32 * 64 / 8, 256, 0, stream>>>(w_out16, ao, out_b, gate, x, outp);
}
